// MHSA_764504179375
// MI455X (gfx1250) — matrix operand order repaired and checked by reading, not compiled, not run on hardware
//
#include <hip/hip_runtime.h>
#include <hip/hip_bf16.h>

// ---------------------------------------------------------------------------
// MHSA forward, B=4, S=2048, D=1024, H=16, DH=64, gfx1250 (MI455X).
// bf16 WMMA (v_wmma_f32_16x16x32_bf16) + async global->LDS double buffering.
// ---------------------------------------------------------------------------

typedef __attribute__((ext_vector_type(16))) __bf16 v16bf;
typedef __attribute__((ext_vector_type(8)))  __bf16 v8bf;
typedef __attribute__((ext_vector_type(4)))  __bf16 v4bf;
typedef __attribute__((ext_vector_type(8)))  float  v8f;

#define WMMA_BF16(A, B, C) \
  __builtin_amdgcn_wmma_f32_16x16x32_bf16(false, (A), false, (B), (short)0, (C), false, false)

// Async global->LDS copy, 16B per lane, tracked by ASYNCcnt (CDNA5 path).
#define ASYNC_LDS_B128(ldsOff, gptr)                                   \
  asm volatile("global_load_async_to_lds_b128 %0, %1, off"             \
               :: "v"((unsigned)(ldsOff)),                             \
                  "v"((unsigned long long)(uintptr_t)(gptr))           \
               : "memory")
#define WAIT_ASYNC0() asm volatile("s_wait_asynccnt 0" ::: "memory")

__device__ __forceinline__ unsigned lds_off(const __bf16* p) {
  return (unsigned)(uintptr_t)p;   // addrspace(3) ptr -> byte offset in LDS
}

// Assemble a 16-element fragment from two 16B chunks.
__device__ __forceinline__ v16bf mkfrag(v8bf lo, v8bf hi) {
  v16bf r;
#pragma unroll
  for (int i = 0; i < 8; ++i) { r[i] = lo[i]; r[i + 8] = hi[i]; }
  return r;
}
// A-fragment: 16 contiguous elements (k = koff..koff+15)
__device__ __forceinline__ v16bf frag_split(const __bf16* p) {
  return mkfrag(*(const v8bf*)p, *(const v8bf*)(p + 8));
}
// B-fragment: 16 contiguous elements (k = koff..koff+15)
__device__ __forceinline__ v16bf frag_cont(const __bf16* p) {
  return mkfrag(*(const v8bf*)p, *(const v8bf*)(p + 8));
}

// ---------------------------------------------------------------------------
// f32 -> bf16 (4 elems/thread)
// ---------------------------------------------------------------------------
__global__ __launch_bounds__(256) void cvt_f32_to_bf16(
    const float* __restrict__ in, __bf16* __restrict__ out, int n4) {
  int i = blockIdx.x * 256 + threadIdx.x;
  if (i < n4) {
    float4 f = ((const float4*)in)[i];
    v4bf o;
    o[0] = (__bf16)f.x; o[1] = (__bf16)f.y; o[2] = (__bf16)f.z; o[3] = (__bf16)f.w;
    ((v4bf*)out)[i] = o;
  }
}

// ---------------------------------------------------------------------------
// f32 [1024,1024] -> bf16 transposed [n][k] (LDS-tiled, both sides coalesced)
// ---------------------------------------------------------------------------
__global__ __launch_bounds__(256) void cvt_transpose_bf16(
    const float* __restrict__ in, __bf16* __restrict__ out) {
  __shared__ float tile[32][33];
  const int tx = threadIdx.x & 31, ty = threadIdx.x >> 5;  // 32 x 8
  const int bx = blockIdx.x, by = blockIdx.y;              // n-tile, k-tile
#pragma unroll
  for (int j = 0; j < 4; ++j)
    tile[ty + j * 8][tx] = in[(size_t)(by * 32 + ty + j * 8) * 1024 + bx * 32 + tx];
  __syncthreads();
#pragma unroll
  for (int j = 0; j < 4; ++j)
    out[(size_t)(bx * 32 + ty + j * 8) * 1024 + by * 32 + tx] =
        (__bf16)tile[tx][ty + j * 8];
}

// ---------------------------------------------------------------------------
// Tiled bf16 GEMM: C[8192,1024] = A[8192,1024] x W (W given TRANSPOSED [n][k]).
// Tile 128(M) x 128(N), BK=32, double-buffered LDS fed by async copies.
// 8 waves in 4(M) x 2(N); each wave 2x4 WMMA tiles -> 8 wmma / barrier.
//   mode 0: bf16 out, head-major S-major [B,H,S,DH]   (Q, K)
//   mode 1: bf16 out, head-major D-major [B,H,DH,S]   (V)
//   mode 2: f32 out, row-major, + residual            (out proj)
// ---------------------------------------------------------------------------
__global__ __launch_bounds__(256) void gemm_bf16(
    const __bf16* __restrict__ Ag,   // [8192,1024]
    const __bf16* __restrict__ Wt,   // [1024,1024] transposed (n-major)
    __bf16* __restrict__ outH,
    float* __restrict__ outF,
    const float* __restrict__ xres,
    int mode) {
  __shared__ __bf16 As[2][128 * 40];   // [m][k] stride 40 (80B, 16B-aligned)
  __shared__ __bf16 Bs[2][128 * 40];   // [n][k] stride 40

  const int bn = blockIdx.x;           // 0..7   (N/128)
  const int bm = blockIdx.y;           // 0..63  (M/128)
  const int tid = threadIdx.x;
  const int lane = tid & 31, w = tid >> 5;
  const int wm = w >> 1, wn = w & 1;
  const int lane15 = lane & 15, laneh = lane >> 4;
  const int koffA = laneh * 16, koffB = laneh * 16;

  const int srow = tid >> 1;           // 0..127 (stage row, used for A and Bs)
  const int scol = (tid & 1) * 16;     // 0 / 16

  v8f acc[2][4];
#pragma unroll
  for (int i = 0; i < 2; ++i)
#pragma unroll
    for (int j = 0; j < 4; ++j)
#pragma unroll
      for (int e = 0; e < 8; ++e) acc[i][j][e] = 0.0f;

  const __bf16* aRow = Ag + (size_t)(bm * 128 + srow) * 1024 + scol;
  const __bf16* wRow = Wt + (size_t)(bn * 128 + srow) * 1024 + scol;

  auto stage = [&](int kk, int buf) {
    unsigned la = lds_off(&As[buf][srow * 40 + scol]);
    unsigned lb = lds_off(&Bs[buf][srow * 40 + scol]);
    ASYNC_LDS_B128(la,      aRow + kk);
    ASYNC_LDS_B128(la + 16, aRow + kk + 8);
    ASYNC_LDS_B128(lb,      wRow + kk);
    ASYNC_LDS_B128(lb + 16, wRow + kk + 8);
  };

  stage(0, 0);
  for (int kk = 0; kk < 1024; kk += 32) {
    const int buf = (kk >> 5) & 1;
    WAIT_ASYNC0();                 // this buffer's tile has landed in LDS
    __syncthreads();               // everyone's tile landed; prior reads drained
    if (kk + 32 < 1024) stage(kk + 32, buf ^ 1);

    v16bf af[2], bfg[4];
#pragma unroll
    for (int s = 0; s < 2; ++s)
      af[s] = frag_split(&As[buf][(wm * 32 + s * 16 + lane15) * 40 + koffA]);
#pragma unroll
    for (int j = 0; j < 4; ++j)
      bfg[j] = frag_cont(&Bs[buf][(wn * 64 + j * 16 + lane15) * 40 + koffB]);
#pragma unroll
    for (int s = 0; s < 2; ++s)
#pragma unroll
      for (int j = 0; j < 4; ++j)
        acc[s][j] = WMMA_BF16(af[s], bfg[j], acc[s][j]);
  }

  // epilogue (C layout: row = r + 8*laneh, col = lane15)
#pragma unroll
  for (int i = 0; i < 2; ++i)
#pragma unroll
    for (int j = 0; j < 4; ++j)
#pragma unroll
      for (int r = 0; r < 8; ++r) {
        const int mg = bm * 128 + wm * 32 + i * 16 + r + laneh * 8;  // 0..8191
        const int ng = bn * 128 + wn * 64 + j * 16 + lane15;         // 0..1023
        const int bb = mg >> 11, sq = mg & 2047;
        const int hh = ng >> 6,  dh = ng & 63;
        if (mode == 0) {
          outH[(((size_t)(bb * 16 + hh) * 2048 + sq) << 6) + dh] = (__bf16)acc[i][j][r];
        } else if (mode == 1) {
          outH[(((size_t)(bb * 16 + hh) * 64 + dh) << 11) + sq] = (__bf16)acc[i][j][r];
        } else {
          const size_t o = (size_t)mg * 1024 + ng;
          outF[o] = xres[o] + acc[i][j][r];
        }
      }
}

// ---------------------------------------------------------------------------
// Flash attention per (b, h, 64-row q-block). 128 threads (4 waves).
// K [B,H,S,64] row-major, V [B,H,64,S] d-major -> both tiles are pure
// async b128 copies. Double-buffered K/V LDS, online softmax, P via LDS.
// ---------------------------------------------------------------------------
__global__ __launch_bounds__(128) void flash_attn(
    const __bf16* __restrict__ qh,   // [B,H,S,64]
    const __bf16* __restrict__ kh,   // [B,H,S,64]
    const __bf16* __restrict__ vh,   // [B,H,64,S]
    __bf16* __restrict__ zb) {       // [B,S,1024]
  __shared__ __bf16 Ks[2][32 * 72];  // [key][d]  stride 72 (144B)
  __shared__ __bf16 Vt[2][64 * 40];  // [d][key]  stride 40 (80B)
  __shared__ __bf16 Ps[4][16 * 40];  // per-wave P [m][key]

  const int qblk = blockIdx.x, h = blockIdx.y, b = blockIdx.z;
  const int tid = threadIdx.x;
  const int w = tid >> 5, lane = tid & 31;
  const int lane15 = lane & 15, laneh = lane >> 4;
  const int koffA = laneh * 16, koffB = laneh * 16;

  const size_t headOff = (size_t)(b * 16 + h) * 2048 * 64;
  const __bf16* Q = qh + headOff;
  const __bf16* K = kh + headOff;
  const __bf16* V = vh + headOff;   // [64][2048]

  // persistent Q fragments
  const int qrow = qblk * 64 + w * 16 + lane15;
  v16bf qf[2];
#pragma unroll
  for (int f = 0; f < 2; ++f)
    qf[f] = frag_split(Q + (size_t)qrow * 64 + f * 32 + koffA);

  float mrun[8], lrun[8];
  v8f acc[4];
#pragma unroll
  for (int r = 0; r < 8; ++r) { mrun[r] = -3.0e38f; lrun[r] = 0.0f; }
#pragma unroll
  for (int t = 0; t < 4; ++t)
#pragma unroll
    for (int e = 0; e < 8; ++e) acc[t][e] = 0.0f;

  const float scale = 0.125f;        // 1/sqrt(64)

  const int skey = tid >> 2, kc0 = (tid & 3) * 16;  // K stage: key row, d chunk
  const int drow = tid >> 1, vc0 = (tid & 1) * 16;  // V stage: d row, key chunk

  auto stage = [&](int kb, int buf) {
    const __bf16* kp = K + (size_t)(kb * 32 + skey) * 64 + kc0;
    unsigned lk = lds_off(&Ks[buf][skey * 72 + kc0]);
    ASYNC_LDS_B128(lk,      kp);
    ASYNC_LDS_B128(lk + 16, kp + 8);
    const __bf16* vp = V + (size_t)drow * 2048 + kb * 32 + vc0;
    unsigned lv = lds_off(&Vt[buf][drow * 40 + vc0]);
    ASYNC_LDS_B128(lv,      vp);
    ASYNC_LDS_B128(lv + 16, vp + 8);
  };

  stage(0, 0);
  for (int kb = 0; kb < 64; ++kb) {
    const int buf = kb & 1;
    WAIT_ASYNC0();
    __syncthreads();
    if (kb + 1 < 64) stage(kb + 1, buf ^ 1);

    // scores: S[16q x 32key], K-dim = 64 (2 wmma per 16-key tile)
    v8f s0, s1;
#pragma unroll
    for (int e = 0; e < 8; ++e) { s0[e] = 0.0f; s1[e] = 0.0f; }
#pragma unroll
    for (int f = 0; f < 2; ++f) {
      const int d0 = f * 32 + koffB;
      v16bf kb0 = frag_cont(&Ks[buf][lane15 * 72 + d0]);
      v16bf kb1 = frag_cont(&Ks[buf][(16 + lane15) * 72 + d0]);
      s0 = WMMA_BF16(qf[f], kb0, s0);
      s1 = WMMA_BF16(qf[f], kb1, s1);
    }

    // online softmax (row reductions inside 16-lane halves)
    float p0[8], p1[8];
#pragma unroll
    for (int r = 0; r < 8; ++r) {
      const float x0 = s0[r] * scale, x1 = s1[r] * scale;
      float rm = fmaxf(x0, x1);
#pragma unroll
      for (int off = 1; off < 16; off <<= 1) rm = fmaxf(rm, __shfl_xor(rm, off, 32));
      const float mnew = fmaxf(mrun[r], rm);
      const float e0 = __expf(x0 - mnew), e1 = __expf(x1 - mnew);
      const float corr = __expf(mrun[r] - mnew);
      float rs = e0 + e1;
#pragma unroll
      for (int off = 1; off < 16; off <<= 1) rs += __shfl_xor(rs, off, 32);
      lrun[r] = lrun[r] * corr + rs;
      mrun[r] = mnew;
#pragma unroll
      for (int t = 0; t < 4; ++t) acc[t][r] *= corr;
      p0[r] = e0; p1[r] = e1;
    }

    // relayout P (C-format -> A-format) through LDS
#pragma unroll
    for (int r = 0; r < 8; ++r) {
      const int M = r + laneh * 8;
      Ps[w][M * 40 + lane15]      = (__bf16)p0[r];
      Ps[w][M * 40 + 16 + lane15] = (__bf16)p1[r];
    }
    __syncthreads();

    // PV: acc[t] += P[16x32] x V[32 x 16t..]
    const v16bf pf = frag_split(&Ps[w][lane15 * 40 + koffA]);
#pragma unroll
    for (int t = 0; t < 4; ++t) {
      v16bf vf = frag_cont(&Vt[buf][(t * 16 + lane15) * 40 + koffB]);
      acc[t] = WMMA_BF16(pf, vf, acc[t]);
    }
  }

  // finalize
#pragma unroll
  for (int r = 0; r < 8; ++r) {
    const float inv = 1.0f / lrun[r];
    const int M = r + laneh * 8;
    const size_t row = (size_t)b * 2048 * 1024 +
                       (size_t)(qblk * 64 + w * 16 + M) * 1024 + h * 64;
#pragma unroll
    for (int t = 0; t < 4; ++t)
      zb[row + t * 16 + lane15] = (__bf16)(acc[t][r] * inv);
  }
}

// ---------------------------------------------------------------------------
// Host launch
// ---------------------------------------------------------------------------
extern "C" void kernel_launch(void* const* d_in, const int* in_sizes, int n_in,
                              void* d_out, int out_size, void* d_ws, size_t ws_size,
                              hipStream_t stream) {
  const float* x    = (const float*)d_in[0];
  // d_in[1] = mask (unused by the reference forward)
  const float* Wq   = (const float*)d_in[2];
  const float* Wk   = (const float*)d_in[3];
  const float* Wv   = (const float*)d_in[4];
  const float* Wout = (const float*)d_in[5];
  float* out = (float*)d_out;

  __bf16* wsb = (__bf16*)d_ws;
  const size_t XB  = 0;                 // x bf16                : 8388608
  const size_t WQB = XB  + 8388608;     // Wq^T bf16             : 1048576
  const size_t WKB = WQB + 1048576;
  const size_t WVB = WKB + 1048576;
  const size_t WOB = WVB + 1048576;
  const size_t QH  = WOB + 1048576;     // q [B,H,S,DH]
  const size_t KH  = QH  + 8388608;     // k [B,H,S,DH]
  const size_t VH  = KH  + 8388608;     // v [B,H,DH,S]
  const size_t ZB  = VH  + 8388608;     // z [B,S,D]
  // total 46,137,344 bf16 = 88 MB

  // 1) conversions (weights transposed so GEMM B-tiles are contiguous)
  cvt_f32_to_bf16<<<8192, 256, 0, stream>>>(x, wsb + XB, 8388608 / 4);
  dim3 tt(32, 32);
  cvt_transpose_bf16<<<tt, 256, 0, stream>>>(Wq,   wsb + WQB);
  cvt_transpose_bf16<<<tt, 256, 0, stream>>>(Wk,   wsb + WKB);
  cvt_transpose_bf16<<<tt, 256, 0, stream>>>(Wv,   wsb + WVB);
  cvt_transpose_bf16<<<tt, 256, 0, stream>>>(Wout, wsb + WOB);

  // 2) projections
  dim3 gg(8, 64);   // N/128, M/128
  gemm_bf16<<<gg, 256, 0, stream>>>(wsb + XB, wsb + WQB, wsb + QH, nullptr, nullptr, 0);
  gemm_bf16<<<gg, 256, 0, stream>>>(wsb + XB, wsb + WKB, wsb + KH, nullptr, nullptr, 0);
  gemm_bf16<<<gg, 256, 0, stream>>>(wsb + XB, wsb + WVB, wsb + VH, nullptr, nullptr, 1);

  // 3) flash attention
  flash_attn<<<dim3(32, 16, 4), 128, 0, stream>>>(wsb + QH, wsb + KH, wsb + VH, wsb + ZB);

  // 4) output projection + residual (f32)
  gemm_bf16<<<gg, 256, 0, stream>>>(wsb + ZB, wsb + WOB, nullptr, out, x, 2);
}
